// MultiheadTemporalLatentAttention_38001870635008
// MI455X (gfx1250) — matrix operand order repaired and checked by reading, not compiled, not run on hardware
//
#include <hip/hip_runtime.h>
#include <cstdint>

// ---------------- problem constants (fixed shapes) ----------------
constexpr int B_DIM = 2;
constexpr int T_DIM = 2048;
constexpr int E_DIM = 2048;
constexpr int H_DIM = 16;
constexpr int NOPE  = 64;
constexpr int ROPE  = 32;
constexpr int VD    = 64;
constexpr int KVR   = 256;
constexpr int DR    = 2;
constexpr int LOWR  = 64;            // KVR/4
constexpr int QKD   = NOPE + ROPE;   // 96
constexpr int KQK   = KVR + ROPE;    // 288 (feature dim of attention)
constexpr float LOG1E4 = 9.210340371976184f;
constexpr float SCALE  = 0.1020620726159658f; // 96^-0.5

typedef __bf16 v16bf __attribute__((ext_vector_type(16)));
typedef float  v8f   __attribute__((ext_vector_type(8)));

__device__ __forceinline__ v8f wmma_bf16(const v16bf& a, const v16bf& b, const v8f& c) {
    return __builtin_amdgcn_wmma_f32_16x16x32_bf16(false, a, false, b, (short)0, c, false, false);
}

// ---------------- CDNA5 async global->LDS copy (guarded) ----------------
#if __has_builtin(__builtin_amdgcn_global_load_async_to_lds_b128) && \
    __has_builtin(__builtin_amdgcn_s_wait_asynccnt)
#define MLA_ASYNC_LDS 1
#else
#define MLA_ASYNC_LDS 0
#endif

#if MLA_ASYNC_LDS
typedef int mla_v4i __attribute__((__vector_size__(16)));
typedef __attribute__((address_space(1))) mla_v4i mla_g4;
typedef __attribute__((address_space(3))) mla_v4i mla_l4;
#endif

__device__ __forceinline__ void copy16_g2l(const void* g, void* l) {
#if MLA_ASYNC_LDS
    __builtin_amdgcn_global_load_async_to_lds_b128(
        (mla_g4*)(uintptr_t)g,
        (mla_l4*)(uint32_t)(uintptr_t)l, 0, 0);
#else
    *reinterpret_cast<uint4*>(l) = *reinterpret_cast<const uint4*>(g);
#endif
}
__device__ __forceinline__ void copy_wait() {
#if MLA_ASYNC_LDS
    __builtin_amdgcn_s_wait_asynccnt(0);
#endif
}

// =================================================================
// Generic tiled GEMM: C(MxN) = A(MxK) * B(KxN), fp32 in, fp32/bf16 out.
// 256 threads = 8 waves, tile 128x128x32, batched via element strides.
// Register-prefetch double buffering; B tile stored transposed so both
// fragment gathers are contiguous ds_load_b128.
// =================================================================
template<bool TRANS_B, bool STORE_BF16>
__global__ __launch_bounds__(256)
void gemm_wmma(const float* __restrict__ A, const float* __restrict__ Bm, void* __restrict__ Cv,
               int M, int N, int K, int lda, int ldb, int ldc,
               long long strideA, long long strideB, long long strideC)
{
    const int tid  = threadIdx.x;
    const int wave = tid >> 5;
    const int lane = tid & 31;
    const int half = lane >> 4;
    const int ml   = lane & 15;
    const int wm   = wave & 3;   // 4 wave rows
    const int wn   = wave >> 2;  // 2 wave cols

    A  += (size_t)blockIdx.z * strideA;
    Bm += (size_t)blockIdx.z * strideB;

    const int m0 = blockIdx.y * 128;
    const int n0 = blockIdx.x * 128;

    __shared__ __bf16 As[128][40];    // [m][k]
    __shared__ __bf16 BsT[128][40];   // [n][k]  (transposed tile)

    v8f acc[2][4] = {};
    float4 aR[4], bR[4];

    auto loadA = [&](int k0) {
        #pragma unroll
        for (int s = 0; s < 4; ++s) {
            int idx4 = tid + s * 256;          // 1024 float4
            int r  = idx4 >> 3;
            int c  = (idx4 & 7) * 4;
            int gr = m0 + r;
            float4 v = {0.f, 0.f, 0.f, 0.f};
            if (gr < M) v = *reinterpret_cast<const float4*>(A + (size_t)gr * lda + k0 + c);
            aR[s] = v;
        }
    };
    auto loadB = [&](int k0) {
        #pragma unroll
        for (int s = 0; s < 4; ++s) {
            int idx4 = tid + s * 256;
            float4 v = {0.f, 0.f, 0.f, 0.f};
            if (TRANS_B) {
                int n  = idx4 >> 3;
                int kc = (idx4 & 7) * 4;
                int gn = n0 + n;
                if (gn < N) v = *reinterpret_cast<const float4*>(Bm + (size_t)gn * ldb + k0 + kc);
            } else {
                int r = idx4 >> 5;
                int c = (idx4 & 31) * 4;
                if (n0 + c < N) v = *reinterpret_cast<const float4*>(Bm + (size_t)(k0 + r) * ldb + n0 + c);
            }
            bR[s] = v;
        }
    };
    auto storeTiles = [&]() {
        #pragma unroll
        for (int s = 0; s < 4; ++s) {
            int idx4 = tid + s * 256;
            int r  = idx4 >> 3;
            int c  = (idx4 & 7) * 4;
            float av[4] = {aR[s].x, aR[s].y, aR[s].z, aR[s].w};
            #pragma unroll
            for (int j = 0; j < 4; ++j) As[r][c + j] = (__bf16)av[j];
            float bv[4] = {bR[s].x, bR[s].y, bR[s].z, bR[s].w};
            if (TRANS_B) {
                int n  = idx4 >> 3;
                int kc = (idx4 & 7) * 4;
                #pragma unroll
                for (int j = 0; j < 4; ++j) BsT[n][kc + j] = (__bf16)bv[j];
            } else {
                int rr = idx4 >> 5;
                int cc = (idx4 & 31) * 4;
                #pragma unroll
                for (int j = 0; j < 4; ++j) BsT[cc + j][rr] = (__bf16)bv[j];
            }
        }
    };

    loadA(0);
    loadB(0);
    for (int k0 = 0; k0 < K; k0 += 32) {
        __syncthreads();
        storeTiles();
        __syncthreads();
        if (k0 + 32 < K) { loadA(k0 + 32); loadB(k0 + 32); }

        v16bf afrag[2];
        #pragma unroll
        for (int i = 0; i < 2; ++i) {
            int row = wm * 32 + i * 16 + ml;
            #pragma unroll
            for (int e = 0; e < 16; ++e) {
                int k = half * 16 + e;
                afrag[i][e] = As[row][k];
            }
        }
        v16bf bfrag[4];
        #pragma unroll
        for (int j = 0; j < 4; ++j) {
            int col = wn * 64 + j * 16 + ml;
            #pragma unroll
            for (int e = 0; e < 16; ++e) bfrag[j][e] = BsT[col][half * 16 + e];
        }
        #pragma unroll
        for (int i = 0; i < 2; ++i)
            #pragma unroll
            for (int j = 0; j < 4; ++j)
                acc[i][j] = wmma_bf16(afrag[i], bfrag[j], acc[i][j]);
    }

    // ---- store C ----
    #pragma unroll
    for (int i = 0; i < 2; ++i) {
        #pragma unroll
        for (int j = 0; j < 4; ++j) {
            int col = n0 + wn * 64 + j * 16 + ml;
            #pragma unroll
            for (int p = 0; p < 8; ++p) {
                int row = m0 + wm * 32 + i * 16 + p + 8 * half;
                if (row < M && col < N) {
                    size_t off = (size_t)blockIdx.z * strideC + (size_t)row * ldc + col;
                    if (STORE_BF16) ((__bf16*)Cv)[off] = (__bf16)acc[i][j][p];
                    else            ((float*)Cv)[off]  = acc[i][j][p];
                }
            }
        }
    }
}

// =================================================================
// prep_kv: layernorm(kv) -> kv_norm; P2 = kv_norm@fc_p_w+b; rope(k_pe)->Kcat[...,256:288]
// =================================================================
__global__ __launch_bounds__(256)
void prep_kv_kernel(const float* __restrict__ kv_all, const int* __restrict__ position,
                    const float* __restrict__ gamma, const float* __restrict__ beta,
                    const float* __restrict__ fc_p_w, const float* __restrict__ fc_p_b,
                    float* __restrict__ kv_norm, float* __restrict__ P2, __bf16* __restrict__ Kcat)
{
    const int bt  = blockIdx.x;
    const int t   = bt % T_DIM;
    const int tid = threadIdx.x;
    const float* row = kv_all + (size_t)bt * KQK;

    __shared__ float sh[KVR];
    __shared__ float red[256];

    float x = row[tid];
    red[tid] = x;
    __syncthreads();
    for (int o = 128; o; o >>= 1) { if (tid < o) red[tid] += red[tid + o]; __syncthreads(); }
    float mean = red[0] * (1.0f / KVR);
    __syncthreads();
    float d = x - mean;
    red[tid] = d * d;
    __syncthreads();
    for (int o = 128; o; o >>= 1) { if (tid < o) red[tid] += red[tid + o]; __syncthreads(); }
    float var = red[0] * (1.0f / KVR);

    float y = d * rsqrtf(var + 1e-5f) * gamma[tid] + beta[tid];
    kv_norm[(size_t)bt * KVR + tid] = y;
    sh[tid] = y;
    __syncthreads();

    if (tid < LOWR) {
        float acc = fc_p_b[tid];
        #pragma unroll 8
        for (int c = 0; c < KVR; ++c) acc += sh[c] * fc_p_w[c * LOWR + tid];
        P2[(size_t)bt * LOWR + tid] = acc;
    }
    if (tid < ROPE / 2) {
        float pos = (float)position[t];
        float inv = __expf(-(2.0f * tid) * LOG1E4 / (float)ROPE);
        float a = pos * inv, ca = __cosf(a), sa = __sinf(a);
        float xr = row[KVR + 2 * tid], xi = row[KVR + 2 * tid + 1];
        Kcat[(size_t)bt * KQK + KVR + 2 * tid]     = (__bf16)(xr * ca - xi * sa);
        Kcat[(size_t)bt * KQK + KVR + 2 * tid + 1] = (__bf16)(xr * sa + xi * ca);
    }
}

// C2[t] = sinusoid_pe[t/DR] @ fc_c_w + fc_c_b
__global__ __launch_bounds__(256)
void c2_kernel(const float* __restrict__ fc_c_w, const float* __restrict__ fc_c_b,
               float* __restrict__ C2)
{
    const int t = blockIdx.x, tid = threadIdx.x;
    __shared__ float pe[KVR];
    int i = tid >> 1;
    float div = __expf(-(2.0f * i) * LOG1E4 / (float)KVR);
    float pos = (float)(t / DR);
    pe[tid] = (tid & 1) ? __cosf(pos * div) : __sinf(pos * div);
    __syncthreads();
    if (tid < LOWR) {
        float acc = fc_c_b[tid];
        #pragma unroll 8
        for (int k = 0; k < KVR; ++k) acc += pe[k] * fc_c_w[k * LOWR + tid];
        C2[(size_t)t * LOWR + tid] = acc;
    }
}

// kv_t: at most 2 terms (chunk mask with DR=2). Write bf16 into Kcat[...,0:256].
__global__ __launch_bounds__(256)
void kvt_kernel(const float* __restrict__ C2, const float* __restrict__ P2,
                const float* __restrict__ kv_norm, __bf16* __restrict__ Kcat)
{
    const int bt = blockIdx.x;
    const int b = bt / T_DIM, t = bt % T_DIM;
    const int tid = threadIdx.x;
    const int s0 = (t & 1) ? (t - 1) : -1;

    __shared__ float r0[LOWR], r1[LOWR];
    __shared__ float w0s, w1s;

    if (tid < LOWR) {
        float c2 = C2[(size_t)t * LOWR + tid];
        r1[tid] = c2 * P2[(size_t)bt * LOWR + tid];
        r0[tid] = (s0 >= 0) ? c2 * P2[((size_t)b * T_DIM + s0) * LOWR + tid] : 0.f;
    }
    __syncthreads();
    if (tid == 0) {
        float a0 = 0.f, a1 = 0.f;
        for (int j = 0; j < LOWR; ++j) { a1 += r1[j]; a0 += r0[j]; }
        w1s = 1.f / (1.f + __expf(-a1));
        w0s = (s0 >= 0) ? 1.f / (1.f + __expf(-a0)) : 0.f;
    }
    __syncthreads();
    float v = w1s * kv_norm[(size_t)bt * KVR + tid];
    if (s0 >= 0) v += w0s * kv_norm[((size_t)b * T_DIM + s0) * KVR + tid];
    Kcat[(size_t)bt * KQK + tid] = (__bf16)v;
}

// rope(q_pe) -> Qcat[...,256:288] bf16
__global__ __launch_bounds__(256)
void qpe_kernel(const float* __restrict__ q, const int* __restrict__ position,
                __bf16* __restrict__ Qcat)
{
    const int bt = blockIdx.x;
    const int t  = bt % T_DIM;
    const int tid = threadIdx.x;
    const int h = tid >> 4, j = tid & 15;
    float pos = (float)position[t];
    float inv = __expf(-(2.0f * j) * LOG1E4 / (float)ROPE);
    float a = pos * inv, ca = __cosf(a), sa = __sinf(a);
    const float* src = q + (size_t)bt * (H_DIM * QKD) + h * QKD + NOPE;
    float xr = src[2 * j], xi = src[2 * j + 1];
    __bf16* dst = Qcat + ((size_t)bt * H_DIM + h) * KQK + KVR;
    dst[2 * j]     = (__bf16)(xr * ca - xi * sa);
    dst[2 * j + 1] = (__bf16)(xr * sa + xi * ca);
}

// =================================================================
// Flash attention: per (s-tile 64, head, batch). 128 threads = 4 waves,
// each wave owns 16 query rows. Q fragments register-resident; K/V tile
// (32 keys x 288 feats, first 256 cols double as V) double-buffered in
// LDS and staged with async global->LDS copies when available.
// Mask: allowed(s,t) = (t==s) || (t<s && t odd).
// =================================================================
__global__ __launch_bounds__(128)
void flash_attn_kernel(const __bf16* __restrict__ Qcat, const __bf16* __restrict__ Kcat,
                       float* __restrict__ Xout)
{
    const int b  = blockIdx.z;
    const int h  = blockIdx.y;
    const int sb = blockIdx.x * 64;
    const int tid  = threadIdx.x;
    const int wave = tid >> 5;
    const int lane = tid & 31;
    const int half = lane >> 4;
    const int ml   = lane & 15;

    __shared__ __bf16 Ks[2][32][296];
    __shared__ __bf16 Ps[4][16][40];

    // ---- Q row fragments into registers (reused across all K tiles) ----
    v16bf qf[9];
    {
        const int qrow = sb + wave * 16 + ml;
        const __bf16* qbase = Qcat + (((size_t)(b * T_DIM + qrow)) * H_DIM + h) * KQK;
        #pragma unroll
        for (int kc = 0; kc < 9; ++kc) {
            *reinterpret_cast<uint4*>(&qf[kc]) =
                *reinterpret_cast<const uint4*>(qbase + kc * 32 + half * 16);
            *(reinterpret_cast<uint4*>(&qf[kc]) + 1) =
                *reinterpret_cast<const uint4*>(qbase + kc * 32 + half * 16 + 8);
        }
    }

    const __bf16* kbase = Kcat + (size_t)b * T_DIM * KQK;
    auto stage = [&](int buf, int jt) {
        #pragma unroll
        for (int it = 0; it < 9; ++it) {
            int idx = tid + it * 128;        // 32*36 x 16B
            int r = idx / 36, c8 = idx % 36;
            copy16_g2l(kbase + ((size_t)(jt * 32 + r)) * KQK + c8 * 8, &Ks[buf][r][c8 * 8]);
        }
    };

    float mrow[8], lrow[8];
    #pragma unroll
    for (int i = 0; i < 8; ++i) { mrow[i] = -3.0e38f; lrow[i] = 0.f; }
    v8f acc[16] = {};

    const int ntiles = sb / 32 + 2;          // covers t <= sb+63
    stage(0, 0);
    for (int jt = 0; jt < ntiles; ++jt) {
        const int cur = jt & 1;
        copy_wait();
        __syncthreads();                     // tile[cur] ready; prev compute done
        if (jt + 1 < ntiles) stage(cur ^ 1, jt + 1);

        // ---- scores: 16x32 = Q(16x288) * K^T ----
        v8f s0 = {}, s1 = {};
        #pragma unroll
        for (int kc = 0; kc < 9; ++kc) {
            v16bf b0, b1;
            #pragma unroll
            for (int e = 0; e < 16; ++e) {
                int k = kc * 32 + half * 16 + e;
                b0[e] = Ks[cur][ml][k];
                b1[e] = Ks[cur][16 + ml][k];
            }
            s0 = wmma_bf16(qf[kc], b0, s0);
            s1 = wmma_bf16(qf[kc], b1, s1);
        }

        // ---- mask + online softmax ----
        float alpha[8];
        #pragma unroll
        for (int i = 0; i < 8; ++i) {
            int srow = sb + wave * 16 + i + 8 * half;
            int t0 = jt * 32 + ml;
            int t1 = t0 + 16;
            float v0 = s0[i] * SCALE;
            float v1 = s1[i] * SCALE;
            if (!((t0 == srow) || ((t0 < srow) && (t0 & 1)))) v0 = -1.0e30f;
            if (!((t1 == srow) || ((t1 < srow) && (t1 & 1)))) v1 = -1.0e30f;
            float mx = fmaxf(v0, v1);
            #pragma unroll
            for (int off = 1; off < 16; off <<= 1) mx = fmaxf(mx, __shfl_xor(mx, off, 16));
            float mnew = fmaxf(mrow[i], mx);
            alpha[i] = __expf(mrow[i] - mnew);
            float e0 = __expf(v0 - mnew);
            float e1 = __expf(v1 - mnew);
            float sum = e0 + e1;
            #pragma unroll
            for (int off = 1; off < 16; off <<= 1) sum += __shfl_xor(sum, off, 16);
            lrow[i] = lrow[i] * alpha[i] + sum;
            mrow[i] = mnew;
            Ps[wave][i + 8 * half][ml]      = (__bf16)e0;
            Ps[wave][i + 8 * half][16 + ml] = (__bf16)e1;
        }
        #pragma unroll
        for (int f = 0; f < 16; ++f)
            #pragma unroll
            for (int i = 0; i < 8; ++i) acc[f][i] *= alpha[i];
        __syncthreads();                     // Ps visible

        // ---- out += P(16x32) @ V(32x256) ----
        v16bf pa;
        #pragma unroll
        for (int e = 0; e < 16; ++e) {
            int k = half * 16 + e;
            pa[e] = Ps[wave][ml][k];
        }
        #pragma unroll
        for (int f = 0; f < 16; ++f) {
            v16bf vb;
            #pragma unroll
            for (int e = 0; e < 16; ++e) vb[e] = Ks[cur][half * 16 + e][f * 16 + ml];
            acc[f] = wmma_bf16(pa, vb, acc[f]);
        }
    }

    // ---- epilogue: normalize, store (B,T,H,256) fp32 ----
    #pragma unroll
    for (int f = 0; f < 16; ++f) {
        int col = f * 16 + ml;
        #pragma unroll
        for (int i = 0; i < 8; ++i) {
            int srow = sb + wave * 16 + i + 8 * half;
            Xout[(((size_t)(b * T_DIM + srow)) * H_DIM + h) * KVR + col] = acc[f][i] / lrow[i];
        }
    }
}

// =================================================================
extern "C" void kernel_launch(void* const* d_in, const int* in_sizes, int n_in,
                              void* d_out, int out_size, void* d_ws, size_t ws_size,
                              hipStream_t stream)
{
    (void)in_sizes; (void)n_in; (void)out_size; (void)ws_size;
    const float* query  = (const float*)d_in[0];
    const float* key    = (const float*)d_in[1];
    const int*   pos    = (const int*)  d_in[3];
    const float* wq     = (const float*)d_in[4];
    const float* wkv_a  = (const float*)d_in[5];
    const float* kv_g   = (const float*)d_in[6];
    const float* kv_b   = (const float*)d_in[7];
    const float* wkv_b  = (const float*)d_in[8];
    const float* wo     = (const float*)d_in[9];
    const float* fc_c_w = (const float*)d_in[10];
    const float* fc_c_b = (const float*)d_in[11];
    const float* fc_p_w = (const float*)d_in[12];
    const float* fc_p_b = (const float*)d_in[13];
    float* out = (float*)d_out;

    const int BT = B_DIM * T_DIM; // 4096
    char* ws = (char*)d_ws;
    auto carve = [&](size_t bytes) {
        char* p = ws;
        ws += (bytes + 1023) & ~(size_t)1023;
        return p;
    };
    float*  q_ws    = (float*) carve((size_t)BT * H_DIM * QKD * 4);
    float*  kv_all  = (float*) carve((size_t)BT * KQK * 4);
    float*  kv_norm = (float*) carve((size_t)BT * KVR * 4);
    float*  P2      = (float*) carve((size_t)BT * LOWR * 4);
    float*  C2      = (float*) carve((size_t)T_DIM * LOWR * 4);
    __bf16* Kcat    = (__bf16*)carve((size_t)BT * KQK * 2);
    __bf16* Qcat    = (__bf16*)carve((size_t)BT * H_DIM * KQK * 2);
    float*  xattn   = (float*) carve((size_t)BT * H_DIM * KVR * 4);
    float*  xv      = (float*) carve((size_t)BT * H_DIM * VD * 4);

    // 1) q = query @ wq
    gemm_wmma<false, false><<<dim3(12, 32, 1), 256, 0, stream>>>(
        query, wq, q_ws, BT, H_DIM * QKD, E_DIM, E_DIM, H_DIM * QKD, H_DIM * QKD, 0, 0, 0);

    // 2) kv_all = key @ wkv_a
    gemm_wmma<false, false><<<dim3(3, 32, 1), 256, 0, stream>>>(
        key, wkv_a, kv_all, BT, KQK, E_DIM, E_DIM, KQK, KQK, 0, 0, 0);

    // 3) layernorm + P2 + rope(k_pe)
    prep_kv_kernel<<<BT, 256, 0, stream>>>(kv_all, pos, kv_g, kv_b, fc_p_w, fc_p_b,
                                           kv_norm, P2, Kcat);
    // 4) C2 from analytic sinusoid PE
    c2_kernel<<<T_DIM, 256, 0, stream>>>(fc_c_w, fc_c_b, C2);

    // 5) kv_t (sparse chunk mix) -> Kcat[...,0:256]
    kvt_kernel<<<BT, 256, 0, stream>>>(C2, P2, kv_norm, Kcat);

    // 6) q_abs = q_nope @ wb[h,:64,:] per head -> Qcat[...,0:256] bf16
    gemm_wmma<false, true><<<dim3(2, 32, H_DIM), 256, 0, stream>>>(
        q_ws, wkv_b, Qcat, BT, KVR, NOPE,
        H_DIM * QKD, KVR, H_DIM * KQK, QKD, (NOPE + VD) * KVR, KQK);

    // 7) rope(q_pe) -> Qcat[...,256:288]
    qpe_kernel<<<BT, 256, 0, stream>>>(q_ws, pos, Qcat);

    // 8) flash attention -> xattn (B,T,H,256)
    flash_attn_kernel<<<dim3(T_DIM / 64, H_DIM, B_DIM), 128, 0, stream>>>(Qcat, Kcat, xattn);

    // 9) xv = xattn @ wb[h,64:,:]^T per head (TRANS_B)
    gemm_wmma<true, false><<<dim3(1, 32, H_DIM), 256, 0, stream>>>(
        xattn, wkv_b + NOPE * KVR, xv, BT, VD, KVR,
        H_DIM * KVR, KVR, H_DIM * VD, KVR, (NOPE + VD) * KVR, VD);

    // 10) out = xv @ wo
    gemm_wmma<false, false><<<dim3(16, 32, 1), 256, 0, stream>>>(
        xv, wo, out, BT, E_DIM, H_DIM * VD, H_DIM * VD, E_DIM, E_DIM, 0, 0, 0);
}
